// MyHGTConv_5471788335647
// MI455X (gfx1250) — matrix operand order repaired and checked by reading, not compiled, not run on hardware
//
#include <hip/hip_runtime.h>
#include <stdint.h>

#define IN_CH 128
#define HID 128
#define HEADS 4
#define DIM 32
#define NT 8
#define ET 8
#define PADX 136   // padded LDS row stride for x tile (bf16 elems; 272B: bank-spread, 16B-aligned)
#define PADK 136   // padded LDS row stride for transposed weight tile

typedef __attribute__((ext_vector_type(16))) __bf16 v16bf;
typedef __attribute__((ext_vector_type(8)))  float  v8f;

// ---------------- WMMA fragment helpers (CDNA5 bf16 16x16x32 layouts) ----------
// A (16x32, 16-bit): row = lane&15; elems 0..7 -> K = kb + half*8 + j,
// elems 8..15 -> K = kb + 16 + half*8 + j.  Two contiguous 8-elem runs -> 2x b128.
__device__ __forceinline__ v16bf frag_a(const __bf16* row, int kb, int half) {
  v16bf f;
#pragma unroll
  for (int j = 0; j < 8; ++j) f[j] = row[kb + half * 8 + j];
#pragma unroll
  for (int j = 0; j < 8; ++j) f[8 + j] = row[kb + 16 + half * 8 + j];
  return f;
}
// B (32x16, 16-bit), weights stored TRANSPOSED in LDS as [col][k]:
// col = lane&15; elems 0..7 -> K = kb + half*8 + j, elems 8..15 -> K = kb + 16 + half*8 + j -> 2x b128.
__device__ __forceinline__ v16bf frag_b_t(const __bf16* colp, int kb, int half) {
  v16bf f;
#pragma unroll
  for (int j = 0; j < 8; ++j) f[j] = colp[kb + half * 8 + j];
#pragma unroll
  for (int j = 0; j < 8; ++j) f[8 + j] = colp[kb + 16 + half * 8 + j];
  return f;
}

// ---------------- bucketing: histogram / prefix+tilemap / scatter ----------------
__global__ void hist_kernel(const int* __restrict__ ntype, int* __restrict__ meta,
                            int nSrc, int nCtr) {
  __shared__ int hs[NT], hc[NT];
  if (threadIdx.x < NT) { hs[threadIdx.x] = 0; hc[threadIdx.x] = 0; }
  __syncthreads();
  int n = blockIdx.x * blockDim.x + threadIdx.x;
  if (n < nSrc) {
    int t = ntype[n];
    atomicAdd(&hs[t], 1);
    if (n < nCtr) atomicAdd(&hc[t], 1);
  }
  __syncthreads();
  if (threadIdx.x < NT) {
    if (hs[threadIdx.x]) atomicAdd(&meta[0 + threadIdx.x], hs[threadIdx.x]);
    if (hc[threadIdx.x]) atomicAdd(&meta[8 + threadIdx.x], hc[threadIdx.x]);
  }
}

__device__ __forceinline__ int4 make_tile_rec(const int* cnt, const int* base, int g) {
  int rem = g;
  for (int t = 0; t < NT; ++t) {
    int nt = (cnt[t] + 15) >> 4;
    if (rem < nt) {
      int sl = rem * 16;
      int valid = cnt[t] - sl; if (valid > 16) valid = 16;
      return make_int4(t, base[t] + sl, valid, 0);
    }
    rem -= nt;
  }
  return make_int4(0, 0, 0, 0);
}

__global__ void setup_kernel(int* __restrict__ meta, int4* __restrict__ tmapS,
                             int4* __restrict__ tmapC, int ubS, int ubC) {
  __shared__ int scnt[NT], ccnt[NT], sbase[NT], cbase[NT];
  if (threadIdx.x == 0) {
    int as = 0, ac = 0;
    for (int t = 0; t < NT; ++t) {
      scnt[t] = meta[t]; ccnt[t] = meta[8 + t];
      sbase[t] = as; cbase[t] = ac;
      meta[16 + t] = as; meta[24 + t] = ac;  // bases
      meta[32 + t] = as; meta[40 + t] = ac;  // scatter cursors
      as += scnt[t]; ac += ccnt[t];
    }
  }
  __syncthreads();
  for (int g = threadIdx.x; g < ubS; g += blockDim.x) tmapS[g] = make_tile_rec(scnt, sbase, g);
  for (int g = threadIdx.x; g < ubC; g += blockDim.x) tmapC[g] = make_tile_rec(ccnt, cbase, g);
}

__global__ void scatter_kernel(const int* __restrict__ ntype, int* __restrict__ meta,
                               int* __restrict__ orderS, int* __restrict__ orderC,
                               int nSrc, int nCtr) {
  int n = blockIdx.x * blockDim.x + threadIdx.x;
  if (n >= nSrc) return;
  int t = ntype[n];
  int p = atomicAdd(&meta[32 + t], 1);
  orderS[p] = n;
  if (n < nCtr) { int p2 = atomicAdd(&meta[40 + t], 1); orderC[p2] = n; }
}

// ---------------- weight conversion (TRANSPOSED: [nt][c][k]) ---------------------
__global__ void cvt_weights(const float* __restrict__ klin, const float* __restrict__ vlin,
                            __bf16* __restrict__ kb, __bf16* __restrict__ vb, int n) {
  int i = blockIdx.x * blockDim.x + threadIdx.x;   // i = ((nt*128 + k)*128 + c) source index
  if (i >= n) return;
  int c = i & 127, k = (i >> 7) & 127, nt = i >> 14;
  size_t o = ((size_t)nt * 128 + c) * 128 + k;     // transposed dest
  kb[o] = (__bf16)klin[i];
  vb[o] = (__bf16)vlin[i];
}

// Wqa[nt][t][c=h*32+d][k=i] = sum_f q_lin[nt,i,h,f] * a_rel[t,h,d,f]   (transposed)
__global__ void wqa_kernel(const float* __restrict__ qlin, const float* __restrict__ arel,
                           __bf16* __restrict__ W, int total) {
  int i = blockIdx.x * blockDim.x + threadIdx.x;
  if (i >= total) return;
  int hd = i & 127, ii = (i >> 7) & 127, t = (i >> 14) & 7, nt = i >> 17;
  int h = hd >> 5, d = hd & 31;
  const float* q = qlin + (((size_t)nt * 128 + ii) * 4 + h) * 32;
  const float* a = arel + (((size_t)t * 4 + h) * 32 + d) * 32;
  float s = 0.f;
#pragma unroll
  for (int f = 0; f < 32; ++f) s += q[f] * a[f];
  W[(((size_t)nt * 8 + t) * 128 + hd) * 128 + ii] = (__bf16)s;
}

// ---------------- shared GEMM-tile device helpers --------------------------------
__device__ __forceinline__ void load_x_tile(const float* __restrict__ x,
                                            const int* __restrict__ snode,
                                            int valid, int tid, __bf16* sx) {
  for (int q = tid; q < 512; q += 128) {          // 16 rows x 128 cols / 4
    int r = q >> 5, c4 = (q & 31) * 4;
    float4 xv = make_float4(0.f, 0.f, 0.f, 0.f);
    if (r < valid) xv = *(const float4*)(x + (size_t)snode[r] * 128 + c4);
    __bf16* dp = sx + r * PADX + c4;
    dp[0] = (__bf16)xv.x; dp[1] = (__bf16)xv.y; dp[2] = (__bf16)xv.z; dp[3] = (__bf16)xv.w;
  }
}

__device__ __forceinline__ void load_w_tile(const __bf16* __restrict__ wsrc,
                                            int tid, __bf16* sw) {
  // wsrc is pre-transposed [c][k]; copy 16384 bf16 with 16B vectors both sides
  for (int e8 = tid; e8 < 2048; e8 += 128) {
    int e = e8 * 8;
    int c = e >> 7, k = e & 127;
    *(uint4*)(sw + c * PADK + k) = *(const uint4*)(wsrc + e);
  }
}

// Precompute per-lane element offsets (node*128 + n) for the 8 D rows this lane owns.
__device__ __forceinline__ void make_row_off(const int* __restrict__ snode, int half, int n,
                                             size_t* off) {
  int4 s0 = *(const int4*)(snode + half * 8);
  int4 s1 = *(const int4*)(snode + half * 8 + 4);
  off[0] = (size_t)s0.x * 128 + n; off[1] = (size_t)s0.y * 128 + n;
  off[2] = (size_t)s0.z * 128 + n; off[3] = (size_t)s0.w * 128 + n;
  off[4] = (size_t)s1.x * 128 + n; off[5] = (size_t)s1.y * 128 + n;
  off[6] = (size_t)s1.z * 128 + n; off[7] = (size_t)s1.w * 128 + n;
}

__device__ __forceinline__ void store_tile(__bf16* __restrict__ dst, const size_t* off,
                                           int colBase, const v8f& c, int half, int valid) {
  if (valid == 16) {
#pragma unroll
    for (int g = 0; g < 8; ++g) dst[off[g] + colBase] = (__bf16)c[g];
  } else {
#pragma unroll
    for (int g = 0; g < 8; ++g)
      if (g + 8 * half < valid) dst[off[g] + colBase] = (__bf16)c[g];
  }
}

// ---------------- Phase 1: typed linear K,V via WMMA (16 nodes / tile) -----------
__global__ void __launch_bounds__(128) gemm_kv(const float* __restrict__ x,
    const int4* __restrict__ tmap, const int* __restrict__ order,
    const __bf16* __restrict__ wkb, const __bf16* __restrict__ wvb,
    __bf16* __restrict__ Kb, __bf16* __restrict__ Vb) {
  __shared__ __bf16 sx[16 * PADX];
  __shared__ __bf16 sw[128 * PADK];
  __shared__ int snode[16];
  int4 rec = tmap[blockIdx.x];
  int type = rec.x, start = rec.y, valid = rec.z;
  if (valid == 0) return;
  int tid = threadIdx.x;
  if (tid < 16) snode[tid] = (tid < valid) ? order[start + tid] : 0;
  __syncthreads();
  load_x_tile(x, snode, valid, tid, sx);
  load_w_tile(wkb + (size_t)type * 16384, tid, sw);
  __syncthreads();
  int lane = tid & 31, w = tid >> 5, half = lane >> 4, n = lane & 15;
  const __bf16* arow = sx + n * PADX;
  size_t off[8];
  make_row_off(snode, half, n, off);
  v16bf afr[4];
#pragma unroll
  for (int kq = 0; kq < 4; ++kq) afr[kq] = frag_a(arow, kq * 32, half);
#pragma unroll
  for (int ti = 0; ti < 2; ++ti) {
    int colBase = (w * 2 + ti) * 16;
    const __bf16* colp = sw + (colBase + n) * PADK;
    v8f c = {};
#pragma unroll
    for (int kq = 0; kq < 4; ++kq) {
      v16bf bf = frag_b_t(colp, kq * 32, half);
      c = __builtin_amdgcn_wmma_f32_16x16x32_bf16(false, afr[kq], false, bf, (short)0, c, false, false);
    }
    store_tile(Kb, off, colBase, c, half, valid);
  }
  __syncthreads();
  load_w_tile(wvb + (size_t)type * 16384, tid, sw);
  __syncthreads();
#pragma unroll
  for (int ti = 0; ti < 2; ++ti) {
    int colBase = (w * 2 + ti) * 16;
    const __bf16* colp = sw + (colBase + n) * PADK;
    v8f c = {};
#pragma unroll
    for (int kq = 0; kq < 4; ++kq) {
      v16bf bf = frag_b_t(colp, kq * 32, half);
      c = __builtin_amdgcn_wmma_f32_16x16x32_bf16(false, afr[kq], false, bf, (short)0, c, false, false);
    }
    store_tile(Vb, off, colBase, c, half, valid);
  }
}

// ---------------- Phase 2: QA[c,t] = x[c] @ Wqa[nt,t]  (fused q_lin * a_rel^T) ---
__global__ void __launch_bounds__(128) gemm_qa(const float* __restrict__ x,
    const int4* __restrict__ tmap, const int* __restrict__ order,
    const __bf16* __restrict__ Wqa, __bf16* __restrict__ QAb) {
  __shared__ __bf16 sx[16 * PADX];
  __shared__ __bf16 sw[128 * PADK];
  __shared__ int snode[16];
  int4 rec = tmap[blockIdx.x];
  int type = rec.x, start = rec.y, valid = rec.z;
  if (valid == 0) return;
  int t = blockIdx.y;
  int tid = threadIdx.x;
  if (tid < 16) snode[tid] = (tid < valid) ? order[start + tid] : 0;
  __syncthreads();
  load_x_tile(x, snode, valid, tid, sx);
  load_w_tile(Wqa + ((size_t)type * 8 + t) * 16384, tid, sw);
  __syncthreads();
  int lane = tid & 31, w = tid >> 5, half = lane >> 4, n = lane & 15;
  const __bf16* arow = sx + n * PADX;
  // QA element offsets: node*1024 + t*128 + n
  int4 s0 = *(const int4*)(snode + half * 8);
  int4 s1 = *(const int4*)(snode + half * 8 + 4);
  size_t off[8];
  size_t tb = (size_t)t * 128 + n;
  off[0] = (size_t)s0.x * 1024 + tb; off[1] = (size_t)s0.y * 1024 + tb;
  off[2] = (size_t)s0.z * 1024 + tb; off[3] = (size_t)s0.w * 1024 + tb;
  off[4] = (size_t)s1.x * 1024 + tb; off[5] = (size_t)s1.y * 1024 + tb;
  off[6] = (size_t)s1.z * 1024 + tb; off[7] = (size_t)s1.w * 1024 + tb;
  v16bf afr[4];
#pragma unroll
  for (int kq = 0; kq < 4; ++kq) afr[kq] = frag_a(arow, kq * 32, half);
#pragma unroll
  for (int ti = 0; ti < 2; ++ti) {
    int colBase = (w * 2 + ti) * 16;
    const __bf16* colp = sw + (colBase + n) * PADK;
    v8f c = {};
#pragma unroll
    for (int kq = 0; kq < 4; ++kq) {
      v16bf bf = frag_b_t(colp, kq * 32, half);
      c = __builtin_amdgcn_wmma_f32_16x16x32_bf16(false, afr[kq], false, bf, (short)0, c, false, false);
    }
    store_tile(QAb, off, colBase, c, half, valid);
  }
}

// ---------------- Phase 3 helpers -----------------------------------------------
__global__ void dst_fill(const int* __restrict__ ptr, int* __restrict__ dst_id, int nCtr) {
  int c = blockIdx.x * blockDim.x + threadIdx.x;
  if (c >= nCtr) return;
  int b = ptr[c], e = ptr[c + 1];
  for (int i = b; i < e; ++i) dst_id[i] = c;
}

// a[e,h] = (k[src] . QA[dst,t]) * rel[h,t] / sqrt(32). One wave per edge.
__global__ void attn_score(const __bf16* __restrict__ Kb, const __bf16* __restrict__ QAb,
                           const int* __restrict__ idx, const int* __restrict__ etype,
                           const int* __restrict__ dst_id, const float* __restrict__ relp,
                           float* __restrict__ a_ws, int nEdge) {
  int wave = blockIdx.x * (blockDim.x >> 5) + (threadIdx.x >> 5);
  if (wave >= nEdge) return;
  int lane = threadIdx.x & 31;
  int e = wave;
  int src = idx[e], t = etype[e], dst = dst_id[e];
  int h = lane >> 3, f0 = (lane & 7) * 4;
  const __bf16* kp = Kb + (size_t)src * 128 + h * 32 + f0;
  const __bf16* qp = QAb + (size_t)dst * 1024 + (size_t)t * 128 + h * 32 + f0;
  float s = 0.f;
#pragma unroll
  for (int j = 0; j < 4; ++j) s += (float)kp[j] * (float)qp[j];
  s += __shfl_xor(s, 1, 32);
  s += __shfl_xor(s, 2, 32);
  s += __shfl_xor(s, 4, 32);
  if ((lane & 7) == 0) {
    float r = relp[h * 8 + t];
    a_ws[(size_t)e * 4 + h] = s * r * 0.17677669529663687f;  // 1/sqrt(32)
  }
}

// Per-center: softmax over CSR range + per-type accumulation + m_rel contraction.
__global__ void __launch_bounds__(128) aggregate(const __bf16* __restrict__ Vb,
    const float* __restrict__ a_ws, const int* __restrict__ ptr,
    const int* __restrict__ idx, const int* __restrict__ etype,
    const float* __restrict__ mrel, float* __restrict__ out, int nCtr) {
  __shared__ float acc[4][ET * 128];
  int w = threadIdx.x >> 5, lane = threadIdx.x & 31;
  int c = blockIdx.x * 4 + w;
  if (c >= nCtr) return;
  int beg = ptr[c], end = ptr[c + 1];
  int h = lane >> 3, f0 = (lane & 7) * 4;
  float* ac = acc[w];
  if (beg >= end) {
    float4 z = make_float4(0.f, 0.f, 0.f, 0.f);
    *(float4*)(out + (size_t)c * 128 + lane * 4) = z;
    return;
  }
  for (int i = lane; i < ET * 128; i += 32) ac[i] = 0.f;
  // pass 1: per-head max
  float m0 = -3.0e38f, m1 = -3.0e38f, m2 = -3.0e38f, m3 = -3.0e38f;
  for (int e = beg + lane; e < end; e += 32) {
    float4 av = *(const float4*)(a_ws + (size_t)e * 4);
    m0 = fmaxf(m0, av.x); m1 = fmaxf(m1, av.y); m2 = fmaxf(m2, av.z); m3 = fmaxf(m3, av.w);
  }
#pragma unroll
  for (int off = 1; off < 32; off <<= 1) {
    m0 = fmaxf(m0, __shfl_xor(m0, off, 32));
    m1 = fmaxf(m1, __shfl_xor(m1, off, 32));
    m2 = fmaxf(m2, __shfl_xor(m2, off, 32));
    m3 = fmaxf(m3, __shfl_xor(m3, off, 32));
  }
  // pass 2: per-head denom
  float s0 = 0.f, s1 = 0.f, s2 = 0.f, s3 = 0.f;
  for (int e = beg + lane; e < end; e += 32) {
    float4 av = *(const float4*)(a_ws + (size_t)e * 4);
    s0 += __expf(av.x - m0); s1 += __expf(av.y - m1);
    s2 += __expf(av.z - m2); s3 += __expf(av.w - m3);
  }
#pragma unroll
  for (int off = 1; off < 32; off <<= 1) {
    s0 += __shfl_xor(s0, off, 32); s1 += __shfl_xor(s1, off, 32);
    s2 += __shfl_xor(s2, off, 32); s3 += __shfl_xor(s3, off, 32);
  }
  float mh = (h == 0) ? m0 : (h == 1) ? m1 : (h == 2) ? m2 : m3;
  float sh = (h == 0) ? s0 : (h == 1) ? s1 : (h == 2) ? s2 : s3;
  float invh = 1.0f / sh;
  // pass 3: accumulate alpha * v into per-edge-type accumulators
  for (int e = beg; e < end; ++e) {
    int t = etype[e], src = idx[e];
    float aE = a_ws[(size_t)e * 4 + h];
    float alpha = __expf(aE - mh) * invh;
    const __bf16* vp = Vb + (size_t)src * 128 + lane * 4;  // == h*32 + f0
    float* ap = ac + t * 128 + lane * 4;
#pragma unroll
    for (int j = 0; j < 4; ++j) ap[j] += alpha * (float)vp[j];
  }
  // final: out[c,h,f] = sum_t sum_d acc[t,h,d] * m_rel[t,h,d,f]
  float o0 = 0.f, o1 = 0.f, o2 = 0.f, o3 = 0.f;
#pragma unroll
  for (int t = 0; t < ET; ++t) {
    const float* mr = mrel + (((size_t)t * 4 + h) * 32) * 32;
    const float* at = ac + t * 128 + h * 32;
#pragma unroll 4
    for (int d = 0; d < 32; ++d) {
      float av = at[d];
      float4 mv = *(const float4*)(mr + d * 32 + f0);
      o0 += av * mv.x; o1 += av * mv.y; o2 += av * mv.z; o3 += av * mv.w;
    }
  }
  *(float4*)(out + (size_t)c * 128 + lane * 4) = make_float4(o0, o1, o2, o3);
}

// ---------------- host-side launch ----------------------------------------------
static inline size_t alignup(size_t v) { return (v + 255) & ~(size_t)255; }

extern "C" void kernel_launch(void* const* d_in, const int* in_sizes, int n_in,
                              void* d_out, int out_size, void* d_ws, size_t ws_size,
                              hipStream_t stream) {
  const float* x     = (const float*)d_in[0];
  const int*   ptr   = (const int*)d_in[1];
  const int*   idx   = (const int*)d_in[2];
  const int*   ntype = (const int*)d_in[3];
  const int*   etype = (const int*)d_in[4];
  const float* klin  = (const float*)d_in[8];
  const float* qlin  = (const float*)d_in[9];
  const float* vlin  = (const float*)d_in[10];
  const float* arel  = (const float*)d_in[11];
  const float* mrel  = (const float*)d_in[12];
  const float* relp  = (const float*)d_in[13];
  float* out = (float*)d_out;

  int nSrc  = in_sizes[3];      // node_type length
  int nCtr  = in_sizes[1] - 1;  // ptr length - 1
  int nEdge = in_sizes[2];      // idx length
  int ubS = nSrc / 16 + NT + 1;
  int ubC = nCtr / 16 + NT + 1;

  uint8_t* p = (uint8_t*)d_ws;
  __bf16* Kb   = (__bf16*)p; p += alignup((size_t)nSrc * 128 * 2);
  __bf16* Vb   = (__bf16*)p; p += alignup((size_t)nSrc * 128 * 2);
  __bf16* QAb  = (__bf16*)p; p += alignup((size_t)nCtr * 1024 * 2);
  __bf16* Wqa  = (__bf16*)p; p += alignup((size_t)NT * ET * 128 * 128 * 2);
  __bf16* klb  = (__bf16*)p; p += alignup((size_t)NT * 128 * 128 * 2);
  __bf16* vlb  = (__bf16*)p; p += alignup((size_t)NT * 128 * 128 * 2);
  float*  a_ws = (float*)p;  p += alignup((size_t)nEdge * 4 * 4);
  int*  dst_id = (int*)p;    p += alignup((size_t)nEdge * 4);
  int*  orderS = (int*)p;    p += alignup((size_t)nSrc * 4);
  int*  orderC = (int*)p;    p += alignup((size_t)nCtr * 4);
  int*  meta   = (int*)p;    p += alignup(4096);
  int4* tmapS  = (int4*)p;   p += alignup((size_t)ubS * sizeof(int4));
  int4* tmapC  = (int4*)p;   p += alignup((size_t)ubC * sizeof(int4));
  (void)ws_size; (void)n_in; (void)out_size;

  hipMemsetAsync(meta, 0, 256, stream);
  hist_kernel<<<(nSrc + 255) / 256, 256, 0, stream>>>(ntype, meta, nSrc, nCtr);
  setup_kernel<<<1, 128, 0, stream>>>(meta, tmapS, tmapC, ubS, ubC);
  scatter_kernel<<<(nSrc + 255) / 256, 256, 0, stream>>>(ntype, meta, orderS, orderC, nSrc, nCtr);
  cvt_weights<<<(NT * 128 * 128 + 255) / 256, 256, 0, stream>>>(klin, vlin, klb, vlb, NT * 128 * 128);
  wqa_kernel<<<(NT * ET * 128 * 128 + 255) / 256, 256, 0, stream>>>(qlin, arel, Wqa, NT * ET * 128 * 128);
  gemm_kv<<<ubS, 128, 0, stream>>>(x, tmapS, orderS, klb, vlb, Kb, Vb);
  gemm_qa<<<dim3(ubC, ET), 128, 0, stream>>>(x, tmapC, orderC, Wqa, QAb);
  dst_fill<<<(nCtr + 255) / 256, 256, 0, stream>>>(ptr, dst_id, nCtr);
  attn_score<<<(nEdge + 7) / 8, 256, 0, stream>>>(Kb, QAb, idx, etype, dst_id, relp, a_ws, nEdge);
  aggregate<<<(nCtr + 3) / 4, 128, 0, stream>>>(Vb, a_ws, ptr, idx, etype, mrel, out, nCtr);
}
